// Informer_90374701842690
// MI455X (gfx1250) — compile-verified, matrix operand order checked by reading, not run on hardware
//
#include <hip/hip_runtime.h>

// ---------------- constants ----------------
#define BB        64
#define LL        512
#define IN_DIM    7
#define DM        512
#define NH        8
#define HD        64
#define NLAYERS   3
#define DFF       2048
#define OUT_WIN   96
#define HEAD_HID  256
#define U_SEL     102     // L // TOP_FACTOR
#define UPAD      112     // U_SEL padded to multiple of 16
#define CHUNK     64      // flash-attention key chunk
#define MTOK      (BB*LL) // 32768

typedef __bf16 bf16_t;
typedef __attribute__((ext_vector_type(16))) __bf16 v16bf;
typedef __attribute__((ext_vector_type(8)))  float  v8f;

#if defined(__has_builtin)
#if __has_builtin(__builtin_amdgcn_tensor_load_to_lds) && \
    __has_builtin(__builtin_amdgcn_s_wait_tensorcnt)
#define USE_TDM 1
#endif
#endif
#ifndef USE_TDM
#define USE_TDM 0
#endif

#if USE_TDM
typedef unsigned tdm_v4u __attribute__((ext_vector_type(4)));
typedef int      tdm_v8i __attribute__((ext_vector_type(8)));
typedef int      tdm_v4i __attribute__((ext_vector_type(4)));

// 2D TDM load: tile (tile_d1 rows x tile_d0 elems of 2 bytes) from a row-major
// tensor (row stride stride_d0 elems) into LDS at lds_addr with optional LDS
// row padding (pad after 16 dwords, pad_amount_dw dwords).
__device__ __forceinline__ void tdm_load_2d_b16(
    unsigned lds_addr, const void* gaddr, unsigned tile_d0, unsigned tile_d1,
    unsigned tensor_d0, unsigned tensor_d1, unsigned stride_d0,
    unsigned pad_interval_enc, unsigned pad_amount_dw) {
  unsigned long long ga = (unsigned long long)gaddr;
  tdm_v4u g0;
  g0[0] = 1u;  // count=1, user descriptor
  g0[1] = lds_addr;
  g0[2] = (unsigned)(ga & 0xFFFFFFFFu);
  g0[3] = (unsigned)((ga >> 32) & 0x01FFFFFFu) | (2u << 30);  // type=2 (image)
  unsigned pad_en = pad_amount_dw ? 1u : 0u;
  unsigned pad_amt = pad_amount_dw ? (pad_amount_dw - 1u) : 0u;
  tdm_v8i g1;
  g1[0] = (int)((1u << 16) | (pad_en << 20) | ((pad_interval_enc & 7u) << 22) |
                ((pad_amt & 0x7Fu) << 25));  // data_size=1 (2B), wg_mask=0
  g1[1] = (int)((tensor_d0 & 0xFFFFu) << 16);
  g1[2] = (int)((tensor_d0 >> 16) | ((tensor_d1 & 0xFFFFu) << 16));
  g1[3] = (int)((tensor_d1 >> 16) | ((tile_d0 & 0xFFFFu) << 16));
  g1[4] = (int)(tile_d1 & 0xFFFFu);  // tile_dim2 = 0
  g1[5] = (int)stride_d0;            // tensor_dim0_stride (low 32)
  g1[6] = 0;                         // stride hi + dim1_stride lo (unused 2D)
  g1[7] = 0;
  tdm_v4i z4 = {0, 0, 0, 0};
  tdm_v8i z8 = {0, 0, 0, 0, 0, 0, 0, 0};
  __builtin_amdgcn_tensor_load_to_lds(g0, g1, z4, z4, z8, 0);
}
#endif

// ---------------- bf16 helpers ----------------
__device__ __forceinline__ bf16_t f2bf(float f) {
  unsigned u = __builtin_bit_cast(unsigned, f);
  unsigned r = (u + 0x7FFFu + ((u >> 16) & 1u)) >> 16;
  return __builtin_bit_cast(bf16_t, (unsigned short)r);
}
__device__ __forceinline__ float bf2f(bf16_t b) {
  unsigned u = ((unsigned)__builtin_bit_cast(unsigned short, b)) << 16;
  return __builtin_bit_cast(float, u);
}

// Load a 16x32 bf16 A-style fragment from row-major storage (stride in elems,
// must be even). CDNA5 layout: lane<16 -> row lr, K pairs {0,1},{2,3},{4,5},{6,7},
// {16..23}; lanes 16..31 shift K by +8.
__device__ __forceinline__ v16bf ldsfragA(const bf16_t* base, int row0, int stride,
                                          int k0, int lane) {
  int lr = lane & 15, lh = (lane >> 4) & 1;
  const bf16_t* p = base + (size_t)(row0 + lr) * stride + k0 + lh * 8;
  v16bf f;
#pragma unroll
  for (int v = 0; v < 8; ++v) {
    int koff = (v < 4) ? 2 * v : 16 + 2 * (v - 4);
    unsigned d = *(const unsigned*)(p + koff);
    f[2 * v]     = __builtin_bit_cast(bf16_t, (unsigned short)(d & 0xFFFFu));
    f[2 * v + 1] = __builtin_bit_cast(bf16_t, (unsigned short)(d >> 16));
  }
  return f;
}

__device__ __forceinline__ v8f wmma_bf16(v16bf a, v16bf b, v8f c) {
  return __builtin_amdgcn_wmma_f32_16x16x32_bf16(false, a, false, b, (short)0, c,
                                                 false, false);
}

// ---------------- generic tiled bf16 GEMM ----------------
// C[M,N] = epilogue(A[M,K] @ W[K,N] + bias), A/W bf16 row-major, fp32 accum.
// Block tile 128x128, 8 waves in 2x4 grid, wave tile 64x32 (4x2 wmma tiles).
// A tile is staged by the Tensor Data Mover when available (wave 0 issues a
// 128x32 2D descriptor with LDS row padding), B tile cooperatively transposed.
template <bool RELU, bool RESID, bool WF32, bool WBF16>
__global__ __launch_bounds__(256) void gemm_bf16_kernel(
    const bf16_t* __restrict__ A, const bf16_t* __restrict__ W,
    const float* __restrict__ bias, const float* __restrict__ resid,
    float* __restrict__ Cf, bf16_t* __restrict__ Cb, int M, int N, int K) {
  __shared__ bf16_t As[128 * 40];  // [row][k], stride 40
  __shared__ bf16_t Bs[128 * 40];  // [col][k], stride 40 (W^T tile)
  const int tid = threadIdx.x;
  const int w = tid >> 5, lane = tid & 31;
  const int lr = lane & 15, lh = (lane >> 4) & 1;
  const int wm = w >> 2, wn = w & 3;  // 2x4 wave grid
  const int bm = blockIdx.y, bn = blockIdx.x;

  v8f acc[4][2];
#pragma unroll
  for (int i = 0; i < 4; ++i)
#pragma unroll
    for (int j = 0; j < 2; ++j)
#pragma unroll
      for (int e = 0; e < 8; ++e) acc[i][j][e] = 0.0f;

  for (int k0 = 0; k0 < K; k0 += 32) {
#if USE_TDM
    // A tile via TDM: 128 rows x 32 bf16, LDS rows padded to 40 elems
    // (16 dwords data -> pad_interval_enc=3, pad 4 dwords -> pad_amount=4).
    if (w == 0) {
      unsigned lds_addr = (unsigned)(size_t)(const void*)As;
      int rows = M - bm * 128;
      if (rows > 128) rows = 128;
      tdm_load_2d_b16(lds_addr, A + (size_t)(bm * 128) * K + k0,
                      /*tile_d0=*/32, /*tile_d1=*/(unsigned)(rows > 0 ? rows : 0),
                      /*tensor_d0=*/(unsigned)(K - k0), /*tensor_d1=*/(unsigned)rows,
                      /*stride_d0=*/(unsigned)K,
                      /*pad_interval_enc=*/3, /*pad_amount_dw=*/4);
    }
#else
    // A tile: 128 rows x 32 k; 2 threads/row, 16 bf16 (two uint4) each
    {
      int ar = tid >> 1, ah = tid & 1;
      int grow = bm * 128 + ar;
      uint4 v0 = {0, 0, 0, 0}, v1 = {0, 0, 0, 0};
      if (grow < M) {
        const uint4* gp = (const uint4*)(A + (size_t)grow * K + k0 + ah * 16);
        v0 = gp[0];
        v1 = gp[1];
        if (k0 + 32 < K) __builtin_prefetch(A + (size_t)grow * K + k0 + 32, 0, 0);
      }
      uint4* sp = (uint4*)(As + ar * 40 + ah * 16);
      sp[0] = v0;
      sp[1] = v1;
    }
#endif
    // B tile (transposed): Bs[n][k] = W[k0+k][bn*128+n]
    {
      int n = tid >> 1, kh = (tid & 1) * 16;
      int gn = bn * 128 + n;
#pragma unroll
      for (int i = 0; i < 16; ++i) {
        int k = kh + i;
        bf16_t v = (gn < N) ? W[(size_t)(k0 + k) * N + gn] : f2bf(0.0f);
        Bs[n * 40 + k] = v;
      }
    }
#if USE_TDM
    if (w == 0) __builtin_amdgcn_s_wait_tensorcnt(0);
#endif
    __syncthreads();

    v16bf af[4], bfb[2];
#pragma unroll
    for (int mt = 0; mt < 4; ++mt) af[mt] = ldsfragA(As, wm * 64 + mt * 16, 40, 0, lane);
#pragma unroll
    for (int nt = 0; nt < 2; ++nt) bfb[nt] = ldsfragA(Bs, wn * 32 + nt * 16, 40, 0, lane);
#pragma unroll
    for (int mt = 0; mt < 4; ++mt)
#pragma unroll
      for (int nt = 0; nt < 2; ++nt) acc[mt][nt] = wmma_bf16(af[mt], bfb[nt], acc[mt][nt]);
    __syncthreads();
  }

  // epilogue: C-layout lane -> col lr, vgpr j -> row j + 8*lh
#pragma unroll
  for (int mt = 0; mt < 4; ++mt)
#pragma unroll
    for (int nt = 0; nt < 2; ++nt)
#pragma unroll
      for (int j = 0; j < 8; ++j) {
        int row = bm * 128 + wm * 64 + mt * 16 + j + 8 * lh;
        int col = bn * 128 + wn * 32 + nt * 16 + lr;
        if (row < M && col < N) {
          float v = acc[mt][nt][j] + bias[col];
          if (RELU) v = fmaxf(v, 0.0f);
          if (RESID) v += resid[(size_t)row * N + col];
          if (WF32) Cf[(size_t)row * N + col] = v;
          if (WBF16) Cb[(size_t)row * N + col] = f2bf(v);
        }
      }
}

// ---------------- embed + PE + LN ----------------
__global__ __launch_bounds__(256) void embed_ln_kernel(
    const float* __restrict__ x, const float* __restrict__ We,
    const float* __restrict__ be, const float* __restrict__ g,
    const float* __restrict__ b, float* __restrict__ h) {
  __shared__ float red[256];
  const int t = blockIdx.x;           // token
  const int tid = threadIdx.x;
  const int pos = t & (LL - 1);
  const float inv_sqrt_d = 0.044194173824159216f;  // 1/sqrt(512)
  float vals[2];
#pragma unroll
  for (int half = 0; half < 2; ++half) {
    int c = tid + half * 256;
    float v = be[c];
#pragma unroll
    for (int i = 0; i < IN_DIM; ++i) v += x[t * IN_DIM + i] * We[i * DM + c];
    v *= inv_sqrt_d;
    int i2 = c & ~1;
    float freq = __expf((float)i2 * (-9.210340371976184f / (float)DM));
    float ang = (float)pos * freq;
    v += (c & 1) ? __cosf(ang) : __sinf(ang);
    vals[half] = v;
  }
  red[tid] = vals[0] + vals[1];
  __syncthreads();
  for (int o = 128; o > 0; o >>= 1) {
    if (tid < o) red[tid] += red[tid + o];
    __syncthreads();
  }
  float mean = red[0] * (1.0f / DM);
  __syncthreads();
  float d0 = vals[0] - mean, d1 = vals[1] - mean;
  red[tid] = d0 * d0 + d1 * d1;
  __syncthreads();
  for (int o = 128; o > 0; o >>= 1) {
    if (tid < o) red[tid] += red[tid + o];
    __syncthreads();
  }
  float inv = rsqrtf(red[0] * (1.0f / DM) + 1e-5f);
  h[(size_t)t * DM + tid]       = d0 * inv * g[tid] + b[tid];
  h[(size_t)t * DM + tid + 256] = d1 * inv * g[tid + 256] + b[tid + 256];
}

// ---------------- LayerNorm -> bf16 (and optional f32) ----------------
template <bool WF32>
__global__ __launch_bounds__(256) void ln_kernel(
    const float* __restrict__ in, const float* __restrict__ g,
    const float* __restrict__ b, float* __restrict__ outf,
    bf16_t* __restrict__ outb) {
  __shared__ float red[256];
  const int t = blockIdx.x;
  const int tid = threadIdx.x;
  float v0 = in[(size_t)t * DM + tid];
  float v1 = in[(size_t)t * DM + tid + 256];
  red[tid] = v0 + v1;
  __syncthreads();
  for (int o = 128; o > 0; o >>= 1) {
    if (tid < o) red[tid] += red[tid + o];
    __syncthreads();
  }
  float mean = red[0] * (1.0f / DM);
  __syncthreads();
  float d0 = v0 - mean, d1 = v1 - mean;
  red[tid] = d0 * d0 + d1 * d1;
  __syncthreads();
  for (int o = 128; o > 0; o >>= 1) {
    if (tid < o) red[tid] += red[tid + o];
    __syncthreads();
  }
  float inv = rsqrtf(red[0] * (1.0f / DM) + 1e-5f);
  float y0 = d0 * inv * g[tid] + b[tid];
  float y1 = d1 * inv * g[tid + 256] + b[tid + 256];
  outb[(size_t)t * DM + tid]       = f2bf(y0);
  outb[(size_t)t * DM + tid + 256] = f2bf(y1);
  if (WF32) {
    outf[(size_t)t * DM + tid]       = y0;
    outf[(size_t)t * DM + tid + 256] = y1;
  }
}

// ---------------- small utility kernels ----------------
__global__ void f32_to_bf16_kernel(const float* __restrict__ src,
                                   bf16_t* __restrict__ dst, long n) {
  long i = (long)blockIdx.x * blockDim.x + threadIdx.x;
  if (i < n) dst[i] = f2bf(src[i]);
}
__global__ void zero_f32_kernel(float* __restrict__ p, long n) {
  long i = (long)blockIdx.x * blockDim.x + threadIdx.x;
  if (i < n) p[i] = 0.0f;
}
__global__ void gather_rep_kernel(const bf16_t* __restrict__ a,
                                  bf16_t* __restrict__ rep) {
  int i = blockIdx.x * blockDim.x + threadIdx.x;  // 64*512
  if (i < BB * DM) {
    int bb = i >> 9, c = i & (DM - 1);
    rep[i] = a[((size_t)bb * LL + (LL - 1)) * DM + c];
  }
}

// ---------------- prob-sparse scoring: scores = qn . mean_k(kn) ----------------
__global__ __launch_bounds__(256) void attn_stats_kernel(
    const bf16_t* __restrict__ QB, const bf16_t* __restrict__ KB,
    float* __restrict__ scores) {
  __shared__ float invn[LL];
  __shared__ float kbar[HD];
  const int bh = blockIdx.x, bb = bh >> 3, hh = bh & 7;
  const int tid = threadIdx.x;
  const bf16_t* kp = KB + ((size_t)bb * LL) * DM + hh * HD;
  const bf16_t* qp = QB + ((size_t)bb * LL) * DM + hh * HD;
  for (int l = tid; l < LL; l += 256) {
    float s = 0.0f;
#pragma unroll 8
    for (int d = 0; d < HD; ++d) {
      float f = bf2f(kp[(size_t)l * DM + d]);
      s += f * f;
    }
    invn[l] = rsqrtf(s);
  }
  __syncthreads();
  if (tid < HD) {
    float acc = 0.0f;
    for (int l = 0; l < LL; ++l) acc += bf2f(kp[(size_t)l * DM + tid]) * invn[l];
    kbar[tid] = acc * (1.0f / LL);
  }
  __syncthreads();
  for (int l = tid; l < LL; l += 256) {
    float dot = 0.0f, s = 0.0f;
#pragma unroll 8
    for (int d = 0; d < HD; ++d) {
      float f = bf2f(qp[(size_t)l * DM + d]);
      dot += f * kbar[d];
      s += f * f;
    }
    scores[(size_t)bh * LL + l] = dot * rsqrtf(s);
  }
}

// ---------------- exact top-k via rank counting ----------------
__global__ __launch_bounds__(512) void topk_kernel(const float* __restrict__ scores,
                                                   int* __restrict__ topidx) {
  __shared__ float s[LL];
  const int bh = blockIdx.x, i = threadIdx.x;
  s[i] = scores[(size_t)bh * LL + i];
  __syncthreads();
  float si = s[i];
  int rank = 0;
  for (int j = 0; j < LL; ++j) {
    float sj = s[j];
    rank += (sj > si) || (sj == si && j < i);
  }
  if (rank < U_SEL) topidx[bh * UPAD + rank] = i;
}

// ---------------- flash-style sparse attention (WMMA) ----------------
// One block per (b,h). Waves 0..6 each own a 16-row tile of the 112 (padded)
// selected queries. Online softmax over 8 chunks of 64 keys.
__global__ __launch_bounds__(256) void attn_kernel(
    const bf16_t* __restrict__ QB, const bf16_t* __restrict__ KB,
    const bf16_t* __restrict__ VB, const int* __restrict__ topidx,
    float* __restrict__ ctx) {
  __shared__ bf16_t qs[UPAD * HD];     // [r][d]
  __shared__ bf16_t ks[CHUNK * HD];    // [k][d]
  __shared__ bf16_t vsT[HD * CHUNK];   // [d][k]
  __shared__ bf16_t pb[UPAD * CHUNK];  // [r][k] probs
  __shared__ float mrow[UPAD], lrow[UPAD];
  const int bh = blockIdx.x, bb = bh >> 3, hh = bh & 7;
  const int tid = threadIdx.x, w = tid >> 5, lane = tid & 31;
  const int lr = lane & 15, lh = (lane >> 4) & 1;

  // stage selected q rows
  for (int r = tid; r < UPAD; r += 256) {
    uint4* dp = (uint4*)(qs + r * HD);
    if (r < U_SEL) {
      int tok = topidx[bh * UPAD + r];
      const uint4* sp = (const uint4*)(QB + ((size_t)bb * LL + tok) * DM + hh * HD);
#pragma unroll
      for (int i = 0; i < 8; ++i) dp[i] = sp[i];
    } else {
      uint4 z = {0, 0, 0, 0};
#pragma unroll
      for (int i = 0; i < 8; ++i) dp[i] = z;
    }
  }
  if (tid < UPAD) {
    mrow[tid] = -3.0e38f;
    lrow[tid] = 0.0f;
  }
  v8f acc[4];
#pragma unroll
  for (int i = 0; i < 4; ++i)
#pragma unroll
    for (int e = 0; e < 8; ++e) acc[i][e] = 0.0f;
  __syncthreads();

  for (int c = 0; c < LL / CHUNK; ++c) {
    // K chunk: [k][d] row-major
    {
      int row = tid >> 2, q4 = tid & 3;
      const uint4* sp =
          (const uint4*)(KB + ((size_t)bb * LL + c * CHUNK + row) * DM + hh * HD + q4 * 16);
      uint4* dp = (uint4*)(ks + row * HD + q4 * 16);
      dp[0] = sp[0];
      dp[1] = sp[1];
    }
    // V chunk transposed: [d][k]
    {
      int d = tid >> 2, q4 = tid & 3;
#pragma unroll
      for (int i = 0; i < 16; ++i) {
        int kr = q4 * 16 + i;
        vsT[d * CHUNK + kr] = VB[((size_t)bb * LL + c * CHUNK + kr) * DM + hh * HD + d];
      }
    }
    __syncthreads();

    if (w < 7) {
      // scores: S = Q_tile (16x64) @ K_chunk^T (64x64)
      v8f s[4];
#pragma unroll
      for (int i = 0; i < 4; ++i)
#pragma unroll
        for (int e = 0; e < 8; ++e) s[i][e] = 0.0f;
#pragma unroll
      for (int kk = 0; kk < 2; ++kk) {
        v16bf aq = ldsfragA(qs, w * 16, HD, kk * 32, lane);
#pragma unroll
        for (int nt = 0; nt < 4; ++nt) {
          v16bf bk = ldsfragA(ks, nt * 16, HD, kk * 32, lane);
          s[nt] = wmma_bf16(aq, bk, s[nt]);
        }
      }
      // online softmax per row; row r = w*16 + j + 8*lh, cols nt*16+lr
#pragma unroll
      for (int j = 0; j < 8; ++j) {
        int r = w * 16 + j + 8 * lh;
        float cmax = -3.0e38f;
#pragma unroll
        for (int nt = 0; nt < 4; ++nt) {
          s[nt][j] *= 0.125f;  // 1/sqrt(HD)
          cmax = fmaxf(cmax, s[nt][j]);
        }
#pragma unroll
        for (int m = 1; m < 16; m <<= 1) cmax = fmaxf(cmax, __shfl_xor(cmax, m, 32));
        float mold = mrow[r];
        float newm = fmaxf(mold, cmax);
        float scale = __expf(mold - newm);
        float psum = 0.0f;
#pragma unroll
        for (int nt = 0; nt < 4; ++nt) {
          float p = __expf(s[nt][j] - newm);
          s[nt][j] = p;
          psum += p;
        }
#pragma unroll
        for (int m = 1; m < 16; m <<= 1) psum += __shfl_xor(psum, m, 32);
        if (lr == 0) {
          lrow[r] = lrow[r] * scale + psum;
          mrow[r] = newm;
        }
#pragma unroll
        for (int ot = 0; ot < 4; ++ot) acc[ot][j] *= scale;
#pragma unroll
        for (int nt = 0; nt < 4; ++nt)
          pb[r * CHUNK + nt * 16 + lr] = f2bf(s[nt][j]);
      }
    }
    __syncthreads();
    if (w < 7) {
      // ctx += P (16x64) @ V_chunk (64x64)
#pragma unroll
      for (int kk = 0; kk < 2; ++kk) {
        v16bf ap = ldsfragA(pb, w * 16, CHUNK, kk * 32, lane);
#pragma unroll
        for (int ot = 0; ot < 4; ++ot) {
          v16bf bv = ldsfragA(vsT, ot * 16, CHUNK, kk * 32, lane);
          acc[ot] = wmma_bf16(ap, bv, acc[ot]);
        }
      }
    }
    __syncthreads();
  }

  if (w < 7) {
#pragma unroll
    for (int j = 0; j < 8; ++j) {
      int r = w * 16 + j + 8 * lh;
      if (r < U_SEL) {
        int tok = topidx[bh * UPAD + r];
        float inv = 1.0f / lrow[r];
#pragma unroll
        for (int ot = 0; ot < 4; ++ot)
          ctx[((size_t)bb * LL + tok) * DM + hh * HD + ot * 16 + lr] = acc[ot][j] * inv;
      }
    }
  }
}

// ---------------- host side ----------------
extern "C" void kernel_launch(void* const* d_in, const int* in_sizes, int n_in,
                              void* d_out, int out_size, void* d_ws, size_t ws_size,
                              hipStream_t stream) {
  (void)in_sizes; (void)n_in; (void)out_size; (void)ws_size;
  auto F = [&](int i) { return (const float*)d_in[i]; };

  // input order: x, embed(W,b), embed_ln(g,b),
  //   per layer: qW,qb,kW,kb,vW,vb,oW,ob, ln1g,ln1b, ln2g,ln2b, ff1W,ff1b, ff2W,ff2b,
  //   enc_norm(g,b), head1(W,b), head2(W,b)
  const float* X = F(0);
  const float* embW = F(1); const float* embB = F(2);
  const float* elng = F(3); const float* elnb = F(4);
  const int PTOP = 5 + 16 * NLAYERS;
  const float* encg = F(PTOP + 0); const float* encb = F(PTOP + 1);
  const float* h1W = F(PTOP + 2);  const float* h1B = F(PTOP + 3);
  const float* h2W = F(PTOP + 4);  const float* h2B = F(PTOP + 5);

  // workspace carve
  size_t off = 0;
  auto carve = [&](size_t bytes) -> void* {
    off = (off + 255) & ~(size_t)255;
    void* p = (char*)d_ws + off;
    off += bytes;
    return p;
  };
  float*  h    = (float*)carve((size_t)MTOK * DM * 4);
  float*  ctx  = (float*)carve((size_t)MTOK * DM * 4);
  bf16_t* abuf = (bf16_t*)carve((size_t)MTOK * DM * 2);
  bf16_t* fbuf = (bf16_t*)carve((size_t)MTOK * DFF * 2);
  bf16_t* qb   = (bf16_t*)carve((size_t)MTOK * DM * 2);
  bf16_t* kb   = (bf16_t*)carve((size_t)MTOK * DM * 2);
  bf16_t* vb   = (bf16_t*)carve((size_t)MTOK * DM * 2);
  float*  scor = (float*)carve((size_t)BB * NH * LL * 4);
  int*    tidx = (int*)carve((size_t)BB * NH * UPAD * 4);
  bf16_t* repb = (bf16_t*)carve((size_t)BB * DM * 2);
  bf16_t* h1b  = (bf16_t*)carve((size_t)BB * HEAD_HID * 2);

  auto cvtW = [&](const float* w, long n) -> bf16_t* {
    bf16_t* dst = (bf16_t*)carve((size_t)n * 2);
    f32_to_bf16_kernel<<<(unsigned)((n + 255) / 256), 256, 0, stream>>>(w, dst, n);
    return dst;
  };

  struct LW {
    bf16_t *q, *k, *v, *o, *f1, *f2;
    const float *qb_, *kb_, *vb_, *ob_, *f1b, *f2b, *g1, *b1, *g2, *b2;
  } lw[NLAYERS];
  for (int l = 0; l < NLAYERS; ++l) {
    int p = 5 + 16 * l;
    lw[l].q = cvtW(F(p + 0), DM * DM);   lw[l].qb_ = F(p + 1);
    lw[l].k = cvtW(F(p + 2), DM * DM);   lw[l].kb_ = F(p + 3);
    lw[l].v = cvtW(F(p + 4), DM * DM);   lw[l].vb_ = F(p + 5);
    lw[l].o = cvtW(F(p + 6), DM * DM);   lw[l].ob_ = F(p + 7);
    lw[l].g1 = F(p + 8);  lw[l].b1 = F(p + 9);
    lw[l].g2 = F(p + 10); lw[l].b2 = F(p + 11);
    lw[l].f1 = cvtW(F(p + 12), DM * DFF); lw[l].f1b = F(p + 13);
    lw[l].f2 = cvtW(F(p + 14), DFF * DM); lw[l].f2b = F(p + 15);
  }
  bf16_t* wh1 = cvtW(h1W, DM * HEAD_HID);
  bf16_t* wh2 = cvtW(h2W, HEAD_HID * OUT_WIN);

  dim3 gproj((DM + 127) / 128, (MTOK + 127) / 128);
  dim3 gff1((DFF + 127) / 128, (MTOK + 127) / 128);

  // embed + PE + LN
  embed_ln_kernel<<<MTOK, 256, 0, stream>>>(X, embW, embB, elng, elnb, h);

  for (int l = 0; l < NLAYERS; ++l) {
    // LN1 -> bf16
    ln_kernel<false><<<MTOK, 256, 0, stream>>>(h, lw[l].g1, lw[l].b1, nullptr, abuf);
    // Q/K/V projections (bf16 out)
    gemm_bf16_kernel<false, false, false, true><<<gproj, 256, 0, stream>>>(
        abuf, lw[l].q, lw[l].qb_, nullptr, nullptr, qb, MTOK, DM, DM);
    gemm_bf16_kernel<false, false, false, true><<<gproj, 256, 0, stream>>>(
        abuf, lw[l].k, lw[l].kb_, nullptr, nullptr, kb, MTOK, DM, DM);
    gemm_bf16_kernel<false, false, false, true><<<gproj, 256, 0, stream>>>(
        abuf, lw[l].v, lw[l].vb_, nullptr, nullptr, vb, MTOK, DM, DM);
    // sparse-query scoring + exact top-k
    attn_stats_kernel<<<BB * NH, 256, 0, stream>>>(qb, kb, scor);
    topk_kernel<<<BB * NH, 512, 0, stream>>>(scor, tidx);
    // zero ctx, run flash attention, convert ctx to bf16
    zero_f32_kernel<<<((long)MTOK * DM + 255) / 256, 256, 0, stream>>>(
        ctx, (long)MTOK * DM);
    attn_kernel<<<BB * NH, 256, 0, stream>>>(qb, kb, vb, tidx, ctx);
    f32_to_bf16_kernel<<<((long)MTOK * DM + 255) / 256, 256, 0, stream>>>(
        ctx, abuf, (long)MTOK * DM);
    // output projection with residual into h
    gemm_bf16_kernel<false, true, true, false><<<gproj, 256, 0, stream>>>(
        abuf, lw[l].o, lw[l].ob_, h, h, nullptr, MTOK, DM, DM);
    // LN2 -> bf16, FF1 (relu, bf16 out), FF2 (+residual into h)
    ln_kernel<false><<<MTOK, 256, 0, stream>>>(h, lw[l].g2, lw[l].b2, nullptr, abuf);
    gemm_bf16_kernel<true, false, false, true><<<gff1, 256, 0, stream>>>(
        abuf, lw[l].f1, lw[l].f1b, nullptr, nullptr, fbuf, MTOK, DFF, DM);
    gemm_bf16_kernel<false, true, true, false><<<gproj, 256, 0, stream>>>(
        fbuf, lw[l].f2, lw[l].f2b, h, h, nullptr, MTOK, DM, DFF);
  }

  // final norm, last-token head
  ln_kernel<false><<<MTOK, 256, 0, stream>>>(h, encg, encb, nullptr, abuf);
  gather_rep_kernel<<<(BB * DM + 255) / 256, 256, 0, stream>>>(abuf, repb);
  dim3 gh1((HEAD_HID + 127) / 128, (BB + 127) / 128);
  gemm_bf16_kernel<true, false, false, true><<<gh1, 256, 0, stream>>>(
      repb, wh1, h1B, nullptr, nullptr, h1b, BB, HEAD_HID, DM);
  dim3 gh2((OUT_WIN + 127) / 128, (BB + 127) / 128);
  gemm_bf16_kernel<false, false, true, false><<<gh2, 256, 0, stream>>>(
      h1b, wh2, h2B, nullptr, (float*)d_out, nullptr, BB, OUT_WIN, HEAD_HID);
}
